// InteractionBlock_85323820302758
// MI455X (gfx1250) — matrix operand order repaired and checked by reading, not compiled, not run on hardware
//
#include <hip/hip_runtime.h>
#include <hip/hip_bf16.h>
#include <cstdint>

// ---------------------------------------------------------------------------
// InteractionBlock for MI455X (gfx1250, wave32, WMMA).
// All dense matmuls use v_wmma_f32_16x16x32_bf16 (bf16 A/B, f32 acc).
// B tiles (shared by all 8 waves of a workgroup) are staged into LDS with
// CDNA5 async global->LDS loads (global_load_async_to_lds_b128, ASYNCcnt),
// double-buffered so WMMAs overlap the next tile's DMA.
// Heavy GEMM (144000 x 832 x 128 wedge MLP) fuses the rho_in gather into the
// A-fragment load.
//
// Input order (setup_inputs dict order; 'params' pytree flattened in jax
// alphabetical order): see comments in kernel_launch.
// Output: [h | X | t] f32 concatenated.
// ---------------------------------------------------------------------------

#define DEV __device__ __forceinline__

typedef __attribute__((ext_vector_type(16))) __bf16        v16bf;
typedef __attribute__((ext_vector_type(8)))  float         v8f;
typedef __attribute__((ext_vector_type(4)))  unsigned int  v4u;

union Frag {
  v16bf v;
  struct { v4u lo; v4u hi; } u;
  unsigned short s[16];
};

constexpr int NN    = 4000;
constexpr int HID   = 128;
constexpr int HEADS = 8;
constexpr int NRBF  = 20;
constexpr int E1C   = 24000;
constexpr int E2C   = 48000;
constexpr int NW    = 144000;   // N*D1*D1 wedges
constexpr int RINP  = 832;      // 829 padded to 832

DEV unsigned short f2bf(float f) {
  unsigned u = __float_as_uint(f);
  u += 0x7FFFu + ((u >> 16) & 1u);     // round-to-nearest-even
  return (unsigned short)(u >> 16);
}
DEV float sigmoidf_(float x) { return 1.0f / (1.0f + __expf(-x)); }
DEV unsigned fenc(float f) { unsigned u = __float_as_uint(f); return (u & 0x80000000u) ? ~u : (u | 0x80000000u); }
DEV float fdec(unsigned u) { return (u & 0x80000000u) ? __uint_as_float(u & 0x7FFFFFFFu) : __uint_as_float(~u); }

// LDS-relative byte address of a pointer into a __shared__ array (strip the
// flat aperture via addrspace(3) cast; async-to-LDS VDST wants this form).
typedef __attribute__((address_space(3))) unsigned short lds_us;
DEV unsigned lds_byte_off(unsigned short* p) {
  return (unsigned)(size_t)(lds_us*)p;
}

// Stage one 32x128 bf16 B tile (cols n0..n0+127, k rows k0..k0+31) into LDS
// using CDNA5 async global->LDS DMA. Thread t copies 32 contiguous bytes:
// col = t/2, half = t&1. Completion tracked with ASYNCcnt.
DEV void stage_b_tile_async(unsigned short* lds_dst,
                            const unsigned short* __restrict__ Bt, int ldb,
                            int n0, int k0, int tid) {
  const int c    = tid >> 1;
  const int half = tid & 1;
  const unsigned short* src = Bt + (size_t)(n0 + c) * ldb + k0 + half * 16;
  unsigned lds_addr = lds_byte_off(lds_dst) + (unsigned)(c * 64 + half * 32);
  asm volatile("global_load_async_to_lds_b128 %0, %1, off\n\t"
               "global_load_async_to_lds_b128 %0, %1, off offset:16"
               :
               : "v"(lds_addr), "v"((unsigned long long)(size_t)src)
               : "memory");
}
DEV void wait_async_all() { asm volatile("s_wait_asynccnt 0x0" ::: "memory"); }

// --------------------------- small utility kernels -------------------------

__global__ void k_fill_zero(float* p, int n) {
  int i = blockIdx.x * blockDim.x + threadIdx.x;
  if (i < n) p[i] = 0.0f;
}

__global__ void k_f32_to_bf16(const float* __restrict__ s, unsigned short* __restrict__ d, int n) {
  int i = blockIdx.x * blockDim.x + threadIdx.x;
  if (i < n) d[i] = f2bf(s[i]);
}

// w [K][N] f32 row-major  ->  out [N][Kpad] bf16 (zero-padded K)
__global__ void k_transpose_bf16(const float* __restrict__ w, unsigned short* __restrict__ out,
                                 int K, int N, int Kpad) {
  int i = blockIdx.x * blockDim.x + threadIdx.x;
  if (i >= N * Kpad) return;
  int n = i / Kpad, k = i % Kpad;
  out[i] = f2bf(k < K ? w[(size_t)k * N + n] : 0.0f);
}

// t_out = t_in + sigmoid(mg) * tanh(tg)
__global__ void k_gate_combine(const float* __restrict__ t_in, const float* __restrict__ mg,
                               const float* __restrict__ tg, float* __restrict__ t_out, int n) {
  int i = blockIdx.x * blockDim.x + threadIdx.x;
  if (i < n) t_out[i] = t_in[i] + sigmoidf_(mg[i]) * tanhf(tg[i]);
}

// cat[e] = [h[src[e]] | h[dst[e]] | t[e]] -> bf16, row stride 384
__global__ void k_cat_gata(const float* __restrict__ h, const float* __restrict__ t,
                           const int* __restrict__ src, const int* __restrict__ dst,
                           unsigned short* __restrict__ cat, int E) {
  int i = blockIdx.x * blockDim.x + threadIdx.x;
  if (i >= E * 384) return;
  int e = i / 384, c = i % 384;
  float v;
  if (c < 128)      v = h[(size_t)src[e] * HID + c];
  else if (c < 256) v = h[(size_t)dst[e] * HID + (c - 128)];
  else              v = t[(size_t)e * HID + (c - 256)];
  cat[i] = f2bf(v);
}

// cat[e] = [h[src1[e]] | h[dst1[e]] | t[e1to2[e]]] -> bf16
__global__ void k_cat_htr(const float* __restrict__ h, const float* __restrict__ t,
                          const int* __restrict__ src, const int* __restrict__ dst,
                          const int* __restrict__ e1to2, unsigned short* __restrict__ cat, int E) {
  int i = blockIdx.x * blockDim.x + threadIdx.x;
  if (i >= E * 384) return;
  int e = i / 384, c = i % 384;
  float v;
  if (c < 128)      v = h[(size_t)src[e] * HID + c];
  else if (c < 256) v = h[(size_t)dst[e] * HID + (c - 128)];
  else              v = t[(size_t)e1to2[e] * HID + (c - 256)];
  cat[i] = f2bf(v);
}

// t[e1to2[e]] += delta[e]   (e1->e2 mapping is injective by construction)
__global__ void k_t_scatter_add(float* __restrict__ t, const float* __restrict__ delta,
                                const int* __restrict__ e1to2, int E) {
  int i = blockIdx.x * blockDim.x + threadIdx.x;
  if (i >= E * HID) return;
  int e = i >> 7, c = i & 127;
  t[(size_t)e1to2[e] * HID + c] += delta[i];
}

__global__ void k_init_red(unsigned* lmax, float* lsum) {
  if (threadIdx.x < HEADS) { lmax[threadIdx.x] = fenc(-3.4e38f); lsum[threadIdx.x] = 0.0f; }
}

// logits[e,hd] = (q.k)/4 + t[e].eg_w[:,hd] + eg_b[hd];  global max via encoded atomicMax
__global__ void k_gata_logits(const float* __restrict__ Hq, const float* __restrict__ Hk,
                              const int* __restrict__ src, const int* __restrict__ dst,
                              const float* __restrict__ t, const float* __restrict__ eg_w,
                              const float* __restrict__ eg_b, float* __restrict__ logits,
                              unsigned* __restrict__ lmax, int E) {
  int i = blockIdx.x * blockDim.x + threadIdx.x;
  if (i >= E * HEADS) return;
  int e = i >> 3, hd = i & 7;
  const float* q = Hq + (size_t)dst[e] * HID + hd * 16;
  const float* k = Hk + (size_t)src[e] * HID + hd * 16;
  float dot = 0.0f;
  #pragma unroll
  for (int j = 0; j < 16; ++j) dot += q[j] * k[j];
  const float* te = t + (size_t)e * HID;
  float tg = 0.0f;
  for (int d = 0; d < HID; ++d) tg += te[d] * eg_w[d * HEADS + hd];
  float lg = dot * 0.25f + tg + eg_b[hd];
  logits[i] = lg;
  atomicMax(&lmax[hd], fenc(lg));
}

__global__ void k_gata_expsum(float* __restrict__ logits, const unsigned* __restrict__ lmax,
                              float* __restrict__ lsum, int E) {
  int i = blockIdx.x * blockDim.x + threadIdx.x;
  if (i >= E * HEADS) return;
  int hd = i & 7;
  float p = __expf(logits[i] - fdec(lmax[hd]));
  logits[i] = p;
  atomicAdd(&lsum[hd], p);
}

// h_up[dst[e]] += alpha[e,hd] * v[src[e]]
__global__ void k_gata_msg(const float* __restrict__ p, const float* __restrict__ lsum,
                           const float* __restrict__ Hv, const int* __restrict__ src,
                           const int* __restrict__ dst, float* __restrict__ h_up, int E) {
  int i = blockIdx.x * blockDim.x + threadIdx.x;
  if (i >= E * HID) return;
  int e = i >> 7, c = i & 127, hd = c >> 4;
  float alpha = p[e * HEADS + hd] / lsum[hd];
  atomicAdd(&h_up[(size_t)dst[e] * HID + c], alpha * Hv[(size_t)src[e] * HID + c]);
}

// ------------------------------ generic WMMA GEMM --------------------------
// C[M x N] = act(A[M x K] @ B[K x N] + bias). A bf16 row-major (lda), Bt bf16
// N-major [N][ldb]. One wave computes a 16x128 tile; block = 8 waves; B tile
// double-buffered through LDS via async DMA (shared across waves). grid.y
// covers N in chunks of 128. mode: 0 store, 1 outF+=, 2 atomicAdd scatter via
// rowmap[m]. act: 0 none, 1 silu. All waves run the full loop (barriers);
// stores guarded.
__global__ __launch_bounds__(256)
void k_gemm_bf16(const unsigned short* __restrict__ A, int lda,
                 const unsigned short* __restrict__ Bt, int ldb,
                 const float* __restrict__ bias,
                 float* outF, unsigned short* outB, int ldo,
                 int M, int K, int act, int mode, const int* __restrict__ rowmap) {
  __shared__ __align__(16) unsigned short ldsB[2][32 * 128];

  const int tid   = threadIdx.x;
  const int wave  = tid >> 5;
  const int lane  = tid & 31;
  const int m0    = (blockIdx.x * 8 + wave) * 16;
  const int n0    = blockIdx.y * 128;
  const int lrow  = lane & 15;
  const int khalf = lane >> 4;
  int arow = m0 + lrow; if (arow >= M) arow = M - 1;

  v8f acc[8] = {};
  const int nkt = K >> 5;

  stage_b_tile_async(ldsB[0], Bt, ldb, n0, 0, tid);
  wait_async_all();
  __syncthreads();

  for (int kt = 0; kt < nkt; ++kt) {
    const int cur = kt & 1;
    if (kt + 1 < nkt) stage_b_tile_async(ldsB[cur ^ 1], Bt, ldb, n0, (kt + 1) << 5, tid);

    Frag fa;
    const unsigned short* pa = A + (size_t)arow * lda + (kt << 5) + khalf * 8;
    fa.u.lo = *(const v4u*)(pa);
    fa.u.hi = *(const v4u*)(pa + 16);

    #pragma unroll
    for (int nt = 0; nt < 8; ++nt) {
      const unsigned short* pb = &ldsB[cur][(nt * 16 + lrow) * 32 + khalf * 8];
      Frag fb;
      fb.u.lo = *(const v4u*)(pb);
      fb.u.hi = *(const v4u*)(pb + 16);
      acc[nt] = __builtin_amdgcn_wmma_f32_16x16x32_bf16(false, fa.v, false, fb.v,
                                                        (short)0, acc[nt], false, false);
    }
    wait_async_all();          // next tile's DMA landed
    __syncthreads();           // everyone done reading cur / writing nxt
  }

  #pragma unroll
  for (int nt = 0; nt < 8; ++nt) {
    const int n  = n0 + nt * 16 + lrow;
    const float bv = bias ? bias[n] : 0.0f;
    #pragma unroll
    for (int r = 0; r < 8; ++r) {
      const int m = m0 + r + khalf * 8;
      if (m >= M) continue;
      float v = acc[nt][r] + bv;
      if (act == 1) v = v * sigmoidf_(v);
      if (outB) outB[(size_t)m * ldo + n] = f2bf(v);
      if (outF) {
        if (mode == 2)      atomicAdd(&outF[(size_t)rowmap[m] * ldo + n], v);
        else if (mode == 1) outF[(size_t)m * ldo + n] += v;
        else                outF[(size_t)m * ldo + n] = v;
      }
    }
  }
}

// --------------------- fused-gather wedge GEMM (rho layer 1) ---------------
// hidden[w] = silu(rho_in[w] @ W1 + b1), rho_in gathered on the fly:
// [t[e1to2[eik]] | t[e1to2[ekj]] | t[eij] | h[wi] | h[wk] | h[wj] |
//  rbf1[eik] | rbf1[ekj] | rbf2[eij] | c_feat | 0 0 0]   (len 832)
// W1 tile double-buffered through LDS via async DMA, shared by all 8 waves.
__global__ __launch_bounds__(256)
void k_rho_gemm1(const unsigned short* __restrict__ te2_bf, const unsigned short* __restrict__ h_bf,
                 const float* __restrict__ rbf1, const float* __restrict__ rbf2,
                 const float* __restrict__ sph1,
                 const int* __restrict__ wi, const int* __restrict__ wk, const int* __restrict__ wj,
                 const int* __restrict__ eik, const int* __restrict__ ekj, const int* __restrict__ eij,
                 const int* __restrict__ e1to2,
                 const unsigned short* __restrict__ W1t, const float* __restrict__ b1,
                 unsigned short* __restrict__ hidden, int Mtot) {
  __shared__ __align__(16) unsigned short ldsB[2][32 * 128];

  const int tid   = threadIdx.x;
  const int wave  = tid >> 5;
  const int lane  = tid & 31;
  const int m0    = (blockIdx.x * 8 + wave) * 16;   // grid sized exactly: always < Mtot
  const int lrow  = lane & 15;
  const int khalf = lane >> 4;
  const int w     = m0 + lrow;

  const int r_eik = eik[w], r_ekj = ekj[w], r_eij = eij[w];
  const unsigned short* seg[6];
  seg[0] = te2_bf + (size_t)e1to2[r_eik] * HID;
  seg[1] = te2_bf + (size_t)e1to2[r_ekj] * HID;
  seg[2] = te2_bf + (size_t)r_eij * HID;
  seg[3] = h_bf   + (size_t)wi[w] * HID;
  seg[4] = h_bf   + (size_t)wk[w] * HID;
  seg[5] = h_bf   + (size_t)wj[w] * HID;

  v8f acc[8] = {};
  constexpr int nkt = RINP / 32;   // 26

  stage_b_tile_async(ldsB[0], W1t, RINP, 0, 0, tid);
  wait_async_all();
  __syncthreads();

  for (int kt = 0; kt < nkt; ++kt) {
    const int cur = kt & 1;
    const int k0  = kt << 5;
    if (kt + 1 < nkt) stage_b_tile_async(ldsB[cur ^ 1], W1t, RINP, 0, (kt + 1) << 5, tid);

    Frag fa;
    if (k0 < 768) {                       // six 128-wide dense segments
      const int ka = k0 + khalf * 8;      // 8-elem chunks never cross 128-bdry
      const int kb = ka + 16;
      fa.u.lo = *(const v4u*)(seg[ka >> 7] + (ka & 127));
      fa.u.hi = *(const v4u*)(seg[kb >> 7] + (kb & 127));
    } else {                              // rbf / c_feat tail, scalar gather
      #pragma unroll
      for (int e = 0; e < 16; ++e) {
        const int k = k0 + (e & 7) + ((e >> 3) << 4) + khalf * 8;
        float v;
        if (k < 788)       v = rbf1[(size_t)r_eik * NRBF + (k - 768)];
        else if (k < 808)  v = rbf1[(size_t)r_ekj * NRBF + (k - 788)];
        else if (k < 828)  v = rbf2[(size_t)r_eij * NRBF + (k - 808)];
        else if (k == 828) v = sph1[(size_t)r_eik * 4 + 1] * sph1[(size_t)r_ekj * 4 + 1];
        else               v = 0.0f;
        fa.s[e] = f2bf(v);
      }
    }
    #pragma unroll
    for (int nt = 0; nt < 8; ++nt) {
      const unsigned short* pb = &ldsB[cur][(nt * 16 + lrow) * 32 + khalf * 8];
      Frag fb;
      fb.u.lo = *(const v4u*)(pb);
      fb.u.hi = *(const v4u*)(pb + 16);
      acc[nt] = __builtin_amdgcn_wmma_f32_16x16x32_bf16(false, fa.v, false, fb.v,
                                                        (short)0, acc[nt], false, false);
    }
    wait_async_all();
    __syncthreads();
  }

  #pragma unroll
  for (int nt = 0; nt < 8; ++nt) {
    const int n = nt * 16 + lrow;
    const float bv = b1[n];
    #pragma unroll
    for (int r = 0; r < 8; ++r) {
      const int m = m0 + r + khalf * 8;
      float v = acc[nt][r] + bv;
      v = v * sigmoidf_(v);               // silu
      hidden[(size_t)m * HID + n] = f2bf(v);
    }
  }
}

// ------------------------------- host side ---------------------------------

static inline void launch_gemm(hipStream_t s, const unsigned short* A, int lda,
                               const unsigned short* Bt, int ldb, const float* bias,
                               float* outF, unsigned short* outB, int ldo,
                               int M, int K, int N, int act, int mode, const int* rowmap) {
  dim3 grid((M + 127) / 128, N / 128);
  k_gemm_bf16<<<grid, 256, 0, s>>>(A, lda, Bt, ldb, bias, outF, outB, ldo, M, K, act, mode, rowmap);
}
static inline int gsz(int n) { return (n + 255) / 256; }

struct GataScratch {
  float *Hq, *Hk, *Hv, *logits, *lsum, *h_up;
  unsigned* lmax;
  unsigned short *hup_bf, *catA, *hid2;
};

static void run_gata(hipStream_t s, float* h, unsigned short* h_bf, float* t,
                     const int* src, const int* dst,
                     const float* q_b, const float* k_b, const float* v_b, const float* o_b,
                     const float* eg_w, const float* eg_b, const float* eu_b1, const float* eu_b2,
                     const unsigned short* wt_q, const unsigned short* wt_k,
                     const unsigned short* wt_v, const unsigned short* wt_o,
                     const unsigned short* wt_eu1, const unsigned short* wt_eu2,
                     const GataScratch& sc) {
  // per-node q/k/v projections (gather commutes with linear)
  launch_gemm(s, h_bf, 128, wt_q, 128, q_b, sc.Hq, nullptr, 128, NN, 128, 128, 0, 0, nullptr);
  launch_gemm(s, h_bf, 128, wt_k, 128, k_b, sc.Hk, nullptr, 128, NN, 128, 128, 0, 0, nullptr);
  launch_gemm(s, h_bf, 128, wt_v, 128, v_b, sc.Hv, nullptr, 128, NN, 128, 128, 0, 0, nullptr);
  k_init_red<<<1, 32, 0, s>>>(sc.lmax, sc.lsum);
  const int nl = E2C * HEADS;
  k_gata_logits<<<gsz(nl), 256, 0, s>>>(sc.Hq, sc.Hk, src, dst, t, eg_w, eg_b, sc.logits, sc.lmax, E2C);
  k_gata_expsum<<<gsz(nl), 256, 0, s>>>(sc.logits, sc.lmax, sc.lsum, E2C);
  const int nz = NN * HID;
  k_fill_zero<<<gsz(nz), 256, 0, s>>>(sc.h_up, nz);
  k_gata_msg<<<gsz(E2C * HID), 256, 0, s>>>(sc.logits, sc.lsum, sc.Hv, src, dst, sc.h_up, E2C);
  k_f32_to_bf16<<<gsz(nz), 256, 0, s>>>(sc.h_up, sc.hup_bf, nz);
  launch_gemm(s, sc.hup_bf, 128, wt_o, 128, o_b, h, nullptr, 128, NN, 128, 128, 0, 1, nullptr); // h +=
  k_f32_to_bf16<<<gsz(nz), 256, 0, s>>>(h, h_bf, nz);
  // edge update MLP: t += mlp2([h_src|h_dst|t])
  k_cat_gata<<<gsz(E2C * 384), 256, 0, s>>>(h, t, src, dst, sc.catA, E2C);
  launch_gemm(s, sc.catA, 384, wt_eu1, 384, eu_b1, nullptr, sc.hid2, 128, E2C, 384, 128, 1, 0, nullptr);
  launch_gemm(s, sc.hid2, 128, wt_eu2, 128, eu_b2, t, nullptr, 128, E2C, 128, 128, 0, 1, nullptr);   // t +=
}

static void run_eqff(hipStream_t s, float* h, unsigned short* h_bf,
                     const float* b1, const float* b2,
                     const unsigned short* wt1, const unsigned short* wt2, unsigned short* hid2) {
  launch_gemm(s, h_bf, 128, wt1, 128, b1, nullptr, hid2, 256, NN, 128, 256, 1, 0, nullptr);
  launch_gemm(s, hid2, 256, wt2, 256, b2, h, nullptr, 128, NN, 256, 128, 0, 1, nullptr); // h +=
  k_f32_to_bf16<<<gsz(NN * HID), 256, 0, s>>>(h, h_bf, NN * HID);
}

extern "C" void kernel_launch(void* const* d_in, const int* in_sizes, int n_in,
                              void* d_out, int out_size, void* d_ws, size_t ws_size,
                              hipStream_t stream) {
  (void)in_sizes; (void)n_in; (void)out_size;
  typedef unsigned short u16;

  // 0 h 1 X 2 t_e2 3 edge_index1 4 edge_index2 5 gata_edge_index 6 gata_to_e2
  // 7 e1_to_e2 8 dist_e1 9 dist_e2 10 rbf_e1 11 rbf_e2 12 sph_e1
  // 13 wedge_i 14 wedge_k 15 wedge_j 16 eik 17 ekj 18 eij 19.. params
  const float* in_h  = (const float*)d_in[0];
  const float* in_X  = (const float*)d_in[1];
  const float* in_t  = (const float*)d_in[2];
  const int*   ei1   = (const int*)d_in[3];            // src1 = ei1, dst1 = ei1+E1C
  const int*   gei   = (const int*)d_in[5];            // src = gei, dst = gei+E2C
  const int*   e1to2 = (const int*)d_in[7];
  const float* rbf1  = (const float*)d_in[10];
  const float* rbf2  = (const float*)d_in[11];
  const float* sph1  = (const float*)d_in[12];
  const int* wi = (const int*)d_in[13], *wk = (const int*)d_in[14], *wj = (const int*)d_in[15];
  const int* eik = (const int*)d_in[16], *ekj = (const int*)d_in[17], *eij = (const int*)d_in[18];

  // param leaves (jax alphabetical pytree order), base 19:
  // e1: b1 b2 w1 w2 | e2: b1 b2 w1 w2 |
  // g1: eg_b eg_w eu_b1 eu_b2 eu_w1 eu_w2 k_b k_w o_b o_w q_b q_w v_b v_w |
  // g2: same | htr: b1 b2 w1 w2 | l2f: gt_b gt_w gw_b gw_w rho_b1 rho_b2 rho_w1 rho_w2
  const float* P[48];
  for (int i = 0; i < 48; ++i) P[i] = (const float*)d_in[19 + i];
  const float *e1_b1 = P[0],  *e1_b2 = P[1],  *e1_w1 = P[2],  *e1_w2 = P[3];
  const float *e2_b1 = P[4],  *e2_b2 = P[5],  *e2_w1 = P[6],  *e2_w2 = P[7];
  const float *g1_egb = P[8],  *g1_egw = P[9],  *g1_eub1 = P[10], *g1_eub2 = P[11];
  const float *g1_euw1 = P[12], *g1_euw2 = P[13], *g1_kb = P[14], *g1_kw = P[15];
  const float *g1_ob = P[16], *g1_ow = P[17], *g1_qb = P[18], *g1_qw = P[19];
  const float *g1_vb = P[20], *g1_vw = P[21];
  const float *g2_egb = P[22], *g2_egw = P[23], *g2_eub1 = P[24], *g2_eub2 = P[25];
  const float *g2_euw1 = P[26], *g2_euw2 = P[27], *g2_kb = P[28], *g2_kw = P[29];
  const float *g2_ob = P[30], *g2_ow = P[31], *g2_qb = P[32], *g2_qw = P[33];
  const float *g2_vb = P[34], *g2_vw = P[35];
  const float *ht_b1 = P[36], *ht_b2 = P[37], *ht_w1 = P[38], *ht_w2 = P[39];
  const float *gt_b = P[40], *gt_w = P[41], *gw_b = P[42], *gw_w = P[43];
  const float *rho_b1 = P[44], *rho_b2 = P[45], *rho_w1 = P[46], *rho_w2 = P[47];

  // ---- workspace arena ----
  size_t off = 0;
  auto alloc = [&](size_t bytes) -> char* {
    off = (off + 255) & ~(size_t)255;
    char* p = (char*)d_ws + off;
    off += bytes;
    return p;
  };
  u16* wt_rho1 = (u16*)alloc((size_t)128 * RINP * 2);
  u16* wt_rho2 = (u16*)alloc((size_t)128 * 128 * 2);
  u16* wt_gw   = (u16*)alloc((size_t)128 * 128 * 2);
  u16* wt_gt   = (u16*)alloc((size_t)128 * 128 * 2);
  u16 *wt_q1 = (u16*)alloc(128*128*2), *wt_k1 = (u16*)alloc(128*128*2);
  u16 *wt_v1 = (u16*)alloc(128*128*2), *wt_o1 = (u16*)alloc(128*128*2);
  u16 *wt_eu1a = (u16*)alloc(128*384*2), *wt_eu2a = (u16*)alloc(128*128*2);
  u16 *wt_q2 = (u16*)alloc(128*128*2), *wt_k2 = (u16*)alloc(128*128*2);
  u16 *wt_v2 = (u16*)alloc(128*128*2), *wt_o2 = (u16*)alloc(128*128*2);
  u16 *wt_eu1b = (u16*)alloc(128*384*2), *wt_eu2b = (u16*)alloc(128*128*2);
  u16 *wt_fw1a = (u16*)alloc(256*128*2), *wt_fw2a = (u16*)alloc(128*256*2);
  u16 *wt_fw1b = (u16*)alloc(256*128*2), *wt_fw2b = (u16*)alloc(128*256*2);
  u16 *wt_hw1  = (u16*)alloc(128*384*2), *wt_hw2  = (u16*)alloc(128*128*2);
  u16* h_bf   = (u16*)alloc((size_t)NN * HID * 2);
  u16* te2_bf = (u16*)alloc((size_t)E2C * HID * 2);
  u16* big_bf = (u16*)alloc((size_t)NW * HID * 2);   // rho hidden, later cat buffers
  u16* hid2   = (u16*)alloc((size_t)E2C * HID * 2);
  u16* hup_bf = (u16*)alloc((size_t)NN * HID * 2);
  u16* m_bf   = (u16*)alloc((size_t)E2C * HID * 2);
  float* m_f32 = (float*)alloc((size_t)E2C * HID * 4);
  float* G1    = (float*)alloc((size_t)E2C * HID * 4);
  float* G2    = (float*)alloc((size_t)E2C * HID * 4);
  float* Hq    = (float*)alloc((size_t)NN * HID * 4);
  float* Hk    = (float*)alloc((size_t)NN * HID * 4);
  float* Hv    = (float*)alloc((size_t)NN * HID * 4);
  float* logits = (float*)alloc((size_t)E2C * HEADS * 4);
  unsigned* lmax = (unsigned*)alloc(HEADS * 4);
  float* lsum    = (float*)alloc(HEADS * 4);
  float* h_up    = (float*)alloc((size_t)NN * HID * 4);
  if (off > ws_size) return;  // workspace too small; nothing safe to do

  float* out_h = (float*)d_out;                    // (4000,128)
  float* out_X = out_h + (size_t)NN * HID;         // (4000,3,128)
  float* out_t = out_X + (size_t)NN * 3 * HID;     // (48000,128)

  // ---- phase 0: pass-throughs, weight prep ----
  hipMemcpyAsync(out_h, in_h, (size_t)NN * HID * 4, hipMemcpyDeviceToDevice, stream);
  hipMemcpyAsync(out_X, in_X, (size_t)NN * 3 * HID * 4, hipMemcpyDeviceToDevice, stream);

  struct TW { const float* src; u16* dst; int K, N, Kpad; };
  const TW tws[] = {
    { rho_w1, wt_rho1, 829, 128, RINP }, { rho_w2, wt_rho2, 128, 128, 128 },
    { gw_w, wt_gw, 128, 128, 128 },      { gt_w, wt_gt, 128, 128, 128 },
    { g1_qw, wt_q1, 128, 128, 128 }, { g1_kw, wt_k1, 128, 128, 128 },
    { g1_vw, wt_v1, 128, 128, 128 }, { g1_ow, wt_o1, 128, 128, 128 },
    { g1_euw1, wt_eu1a, 384, 128, 384 }, { g1_euw2, wt_eu2a, 128, 128, 128 },
    { g2_qw, wt_q2, 128, 128, 128 }, { g2_kw, wt_k2, 128, 128, 128 },
    { g2_vw, wt_v2, 128, 128, 128 }, { g2_ow, wt_o2, 128, 128, 128 },
    { g2_euw1, wt_eu1b, 384, 128, 384 }, { g2_euw2, wt_eu2b, 128, 128, 128 },
    { e1_w1, wt_fw1a, 128, 256, 128 }, { e1_w2, wt_fw2a, 256, 128, 256 },
    { e2_w1, wt_fw1b, 128, 256, 128 }, { e2_w2, wt_fw2b, 256, 128, 256 },
    { ht_w1, wt_hw1, 384, 128, 384 },  { ht_w2, wt_hw2, 128, 128, 128 },
  };
  for (const TW& t : tws) {
    int n = t.N * t.Kpad;
    k_transpose_bf16<<<gsz(n), 256, 0, stream>>>(t.src, t.dst, t.K, t.N, t.Kpad);
  }
  k_f32_to_bf16<<<gsz(NN * HID), 256, 0, stream>>>(in_h, h_bf, NN * HID);
  k_f32_to_bf16<<<gsz(E2C * HID), 256, 0, stream>>>(in_t, te2_bf, E2C * HID);
  k_fill_zero<<<gsz(E2C * HID), 256, 0, stream>>>(m_f32, E2C * HID);

  // ---- phase 1: wedge MLP -> segment_sum -> gating ----
  k_rho_gemm1<<<NW / 128, 256, 0, stream>>>(te2_bf, h_bf, rbf1, rbf2, sph1,
                                            wi, wk, wj, eik, ekj, eij, e1to2,
                                            wt_rho1, rho_b1, big_bf, NW);
  launch_gemm(stream, big_bf, 128, wt_rho2, 128, rho_b2, m_f32, nullptr, 128,
              NW, 128, 128, 0, 2, eij);                       // atomic scatter by eij
  k_f32_to_bf16<<<gsz(E2C * HID), 256, 0, stream>>>(m_f32, m_bf, E2C * HID);
  launch_gemm(stream, m_bf, 128, wt_gw, 128, gw_b, G1, nullptr, 128, E2C, 128, 128, 0, 0, nullptr);
  launch_gemm(stream, te2_bf, 128, wt_gt, 128, gt_b, G2, nullptr, 128, E2C, 128, 128, 0, 0, nullptr);
  k_gate_combine<<<gsz(E2C * HID), 256, 0, stream>>>(in_t, G1, G2, out_t, E2C * HID);

  GataScratch sc{ Hq, Hk, Hv, logits, lsum, h_up, lmax, hup_bf, big_bf, hid2 };
  const int* src2 = gei; const int* dst2 = gei + E2C;

  // ---- phase 2: GATA 1 ----
  run_gata(stream, out_h, h_bf, out_t, src2, dst2,
           g1_qb, g1_kb, g1_vb, g1_ob, g1_egw, g1_egb, g1_eub1, g1_eub2,
           wt_q1, wt_k1, wt_v1, wt_o1, wt_eu1a, wt_eu2a, sc);
  // ---- phase 3: EqFF 1 ----
  run_eqff(stream, out_h, h_bf, e1_b1, e1_b2, wt_fw1a, wt_fw2a, hid2);
  // ---- phase 4: htr edge refinement on E1 ----
  k_cat_htr<<<gsz(E1C * 384), 256, 0, stream>>>(out_h, out_t, ei1, ei1 + E1C, e1to2, big_bf, E1C);
  launch_gemm(stream, big_bf, 384, wt_hw1, 384, ht_b1, nullptr, hid2, 128, E1C, 384, 128, 1, 0, nullptr);
  launch_gemm(stream, hid2, 128, wt_hw2, 128, ht_b2, G1, nullptr, 128, E1C, 128, 128, 0, 0, nullptr);
  k_t_scatter_add<<<gsz(E1C * HID), 256, 0, stream>>>(out_t, G1, e1to2, E1C);
  // ---- phase 5: GATA 2 ----
  run_gata(stream, out_h, h_bf, out_t, src2, dst2,
           g2_qb, g2_kb, g2_vb, g2_ob, g2_egw, g2_egb, g2_eub1, g2_eub2,
           wt_q2, wt_k2, wt_v2, wt_o2, wt_eu1b, wt_eu2b, sc);
  // ---- phase 6: EqFF 2 ----
  run_eqff(stream, out_h, h_bf, e2_b1, e2_b2, wt_fw1b, wt_fw2b, hid2);
}
